// DariushMultiHeadAttention_19533511262804
// MI455X (gfx1250) — matrix operand order repaired and checked by reading, not compiled, not run on hardware
//
#include <hip/hip_runtime.h>

// ---------------- problem constants ----------------
constexpr int S    = 2048;
constexpr int EMB  = 1024;
constexpr int NQ   = 16;
constexpr int NKV  = 4;
constexpr int D    = 64;
constexpr int NQD  = NQ * D;    // 1024
constexpr int NKVD = NKV * D;   // 256

typedef __attribute__((ext_vector_type(16))) __bf16 v16bf;
typedef __attribute__((ext_vector_type(8)))  float  v8f;

// ---------------- bf16 helpers ----------------
__device__ __forceinline__ unsigned short f2bf(float f) {
  union { float f; unsigned u; } c; c.f = f;
  unsigned u = c.u + 0x7FFFu + ((c.u >> 16) & 1u);   // round to nearest even
  return (unsigned short)(u >> 16);
}

// ---------------- WMMA fragments (CDNA5 wave32 layouts) ----------------
union AFrag { v16bf v; unsigned short h[16]; unsigned u[8]; };
union BFrag { v16bf v; unsigned short h[16]; unsigned u[8]; };

// A: 16x32 bf16, row-major source with leading dim `ld`, tile at (row0, k0).
// lane L: m = L%16, half = L/16; VGPR v<4 -> K = 2v + 8*half (+1),
//                               VGPR v>=4 -> K = 16 + 2(v-4) + 8*half (+1).
__device__ __forceinline__ AFrag load_a(const unsigned short* src, int ld,
                                        int row0, int k0, int lane) {
  AFrag a;
  int m = lane & 15, half = lane >> 4;
  const unsigned short* p = src + (size_t)(row0 + m) * ld + k0;
#pragma unroll
  for (int v = 0; v < 8; ++v) {
    int kb = (v < 4 ? 2 * v : 16 + 2 * (v - 4)) + 8 * half;
    a.u[v] = *(const unsigned*)(p + kb);
  }
  return a;
}

// B: 32x16 bf16, B(k,n) = srcT[(n0+n)*ld + k0 + k]  (column-contiguous storage).
// lane L: n = L%16, half = L/16; same K order as load_a (VGPR v<4 -> K = 2v + 8*half, v>=4 -> K = 16 + 2(v-4) + 8*half).
__device__ __forceinline__ BFrag load_bT(const unsigned short* srcT, int ld,
                                         int n0, int k0, int lane) {
  BFrag b;
  int n = lane & 15, half = lane >> 4;
  const unsigned short* p = srcT + (size_t)(n0 + n) * ld + k0;
#pragma unroll
  for (int v = 0; v < 8; ++v) {
    int kb = (v < 4 ? 2 * v : 16 + 2 * (v - 4)) + 8 * half;
    b.u[v] = *(const unsigned*)(p + kb);
  }
  return b;
}

__device__ __forceinline__ v8f wmma_bf16(const AFrag& a, const BFrag& b, v8f c) {
  return __builtin_amdgcn_wmma_f32_16x16x32_bf16(
      /*neg_a=*/false, a.v, /*neg_b=*/false, b.v,
      /*c_mod=*/(short)0, c, /*reuse_a=*/false, /*reuse_b=*/false);
}

// ---------------- precision-cast kernels ----------------
__global__ void cast_f32_bf16_kernel(const float* __restrict__ in,
                                     unsigned short* __restrict__ out, int n) {
  int i = blockIdx.x * blockDim.x + threadIdx.x;
  if (i < n) out[i] = f2bf(in[i]);
}

// in[rows][cols] (fp32) -> out[cols][rows] (bf16)
__global__ void transpose_cast_kernel(const float* __restrict__ in,
                                      unsigned short* __restrict__ out,
                                      int rows, int cols) {
  int i = blockIdx.x * blockDim.x + threadIdx.x;
  if (i < rows * cols) {
    int r = i / cols, c = i % cols;
    out[(size_t)c * rows + r] = f2bf(in[i]);
  }
}

// ---------------- fused QKV projection + bias + RoPE ----------------
// One wave computes a 16(seq) x 64(full head) tile so RoPE pairs (d, d+32)
// live in accumulator blocks nb and nb+2 of the same lane.
// blockIdx.y: 0..15 = Q heads, 16..19 = K heads, 20..23 = V heads.
__global__ __launch_bounds__(32)
void qkv_rope_kernel(const unsigned short* __restrict__ Xbf,   // [S][EMB]
                     const unsigned short* __restrict__ Wqt,   // [NQD][EMB]
                     const float* __restrict__ bq,
                     const unsigned short* __restrict__ Wkt,   // [NKVD][EMB]
                     const float* __restrict__ bk,
                     const unsigned short* __restrict__ Wvt,   // [NKVD][EMB]
                     const float* __restrict__ bv,
                     unsigned short* __restrict__ Qb,          // [NQ][S][D]
                     unsigned short* __restrict__ Kb,          // [NKV][S][D]
                     unsigned short* __restrict__ Vt) {        // [NKV][D][S]
  int lane = threadIdx.x;
  int s0 = blockIdx.x * 16;
  int t = blockIdx.y;

  const unsigned short* Wt; const float* bias; int head, kind;
  if (t < NQ)            { kind = 0; head = t;       Wt = Wqt; bias = bq; }
  else if (t < NQ + NKV) { kind = 1; head = t - NQ;  Wt = Wkt; bias = bk; }
  else                   { kind = 2; head = t - NQ - NKV; Wt = Wvt; bias = bv; }
  int col0 = head * D;

  v8f acc[4] = {};
  for (int k0 = 0; k0 < EMB; k0 += 32) {
    AFrag a = load_a(Xbf, EMB, s0, k0, lane);
#pragma unroll
    for (int nb = 0; nb < 4; ++nb) {
      BFrag b = load_bT(Wt, EMB, col0 + nb * 16, k0, lane);
      acc[nb] = wmma_bf16(a, b, acc[nb]);
    }
  }

  int n = lane & 15, half = lane >> 4;
#pragma unroll
  for (int nb = 0; nb < 4; ++nb) {
    float bb = bias[col0 + nb * 16 + n];
#pragma unroll
    for (int r = 0; r < 8; ++r) acc[nb][r] += bb;
  }

  if (kind == 2) {
    // V: store transposed  Vt[head][d][s]
#pragma unroll
    for (int nb = 0; nb < 4; ++nb) {
      int d = nb * 16 + n;
#pragma unroll
      for (int r = 0; r < 8; ++r) {
        int s = s0 + r + 8 * half;
        Vt[((size_t)head * D + d) * S + s] = f2bf(acc[nb][r]);
      }
    }
    return;
  }

  // RoPE: d < 32 uses partner d+32 (acc block nb+2), angle = s * base^(-d/32)
#pragma unroll
  for (int nb = 0; nb < 2; ++nb) {
    int d = nb * 16 + n;
    float invf = __powf(10000.0f, -(float)d * (1.0f / 32.0f));
#pragma unroll
    for (int r = 0; r < 8; ++r) {
      float pos = (float)(s0 + r + 8 * half);
      float sv, cv;
      __sincosf(pos * invf, &sv, &cv);
      float x1 = acc[nb][r], x2 = acc[nb + 2][r];
      acc[nb][r]     = x1 * cv - x2 * sv;
      acc[nb + 2][r] = x2 * cv + x1 * sv;
    }
  }

  unsigned short* dst = (kind == 0) ? Qb : Kb;
#pragma unroll
  for (int nb = 0; nb < 4; ++nb) {
    int d = nb * 16 + n;
#pragma unroll
    for (int r = 0; r < 8; ++r) {
      int s = s0 + r + 8 * half;
      dst[((size_t)head * S + s) * D + d] = f2bf(acc[nb][r]);
    }
  }
}

// ---------------- causal flash attention (one wave = 16 queries x 1 head) ----
__global__ __launch_bounds__(32)
void attn_kernel(const unsigned short* __restrict__ Qb,  // [NQ][S][D]
                 const unsigned short* __restrict__ Kb,  // [NKV][S][D]
                 const unsigned short* __restrict__ Vt,  // [NKV][D][S]
                 unsigned short* __restrict__ Ob) {      // [S][NQ*D]
  __shared__ unsigned short Plds[16 * 32];               // P tile, row-major
  int lane = threadIdx.x;
  int q0 = blockIdx.x * 16;
  int h  = blockIdx.y;
  int hk = h >> 2;                                       // g = NQ/NKV = 4

  const unsigned short* Qh = Qb + (size_t)h  * S * D;
  const unsigned short* Kh = Kb + (size_t)hk * S * D;
  const unsigned short* Vh = Vt + (size_t)hk * D * S;

  int n = lane & 15, half = lane >> 4;
  const float scale = 0.125f;                            // 1/sqrt(64)

  float mrow[8], lrow[8];
  v8f o[4] = {};
#pragma unroll
  for (int r = 0; r < 8; ++r) { mrow[r] = -1e30f; lrow[r] = 0.0f; }

  AFrag qa0 = load_a(Qh, D, q0, 0, lane);
  AFrag qa1 = load_a(Qh, D, q0, 32, lane);

  for (int kt = 0; kt <= q0; kt += 32) {                 // keys kt .. kt+31
    // ---- S = scale * Q K^T (two 16-key subtiles) ----
    v8f sc[2];
#pragma unroll
    for (int sub = 0; sub < 2; ++sub) {
      int kts = kt + sub * 16;
      v8f sa = {};
      BFrag bk0 = load_bT(Kh, D, kts, 0, lane);
      sa = wmma_bf16(qa0, bk0, sa);
      BFrag bk1 = load_bT(Kh, D, kts, 32, lane);
      sa = wmma_bf16(qa1, bk1, sa);
      sc[sub] = sa;
    }

    // ---- causal mask + per-row max ----
    float rmax[8];
#pragma unroll
    for (int r = 0; r < 8; ++r) rmax[r] = -1e30f;
#pragma unroll
    for (int sub = 0; sub < 2; ++sub) {
      int key = kt + sub * 16 + n;
#pragma unroll
      for (int r = 0; r < 8; ++r) {
        int qi = q0 + r + 8 * half;
        float v = sc[sub][r] * scale;
        v = (key <= qi) ? v : -1e30f;
        sc[sub][r] = v;
        rmax[r] = fmaxf(rmax[r], v);
      }
    }
#pragma unroll
    for (int r = 0; r < 8; ++r) {
      float v = rmax[r];
#pragma unroll
      for (int msk = 1; msk < 16; msk <<= 1)
        v = fmaxf(v, __shfl_xor(v, msk, 32));
      rmax[r] = v;
    }

    // ---- online softmax rescale ----
    float alpha[8];
#pragma unroll
    for (int r = 0; r < 8; ++r) {
      float mnew = fmaxf(mrow[r], rmax[r]);
      alpha[r] = __expf(mrow[r] - mnew);
      mrow[r] = mnew;
    }
#pragma unroll
    for (int nb = 0; nb < 4; ++nb)
#pragma unroll
      for (int r = 0; r < 8; ++r) o[nb][r] *= alpha[r];

    float rsum[8];
#pragma unroll
    for (int r = 0; r < 8; ++r) rsum[r] = 0.0f;
#pragma unroll
    for (int sub = 0; sub < 2; ++sub)
#pragma unroll
      for (int r = 0; r < 8; ++r) {
        float e = __expf(sc[sub][r] - mrow[r]);
        sc[sub][r] = e;
        rsum[r] += e;
      }
#pragma unroll
    for (int r = 0; r < 8; ++r) {
      float v = rsum[r];
#pragma unroll
      for (int msk = 1; msk < 16; msk <<= 1)
        v += __shfl_xor(v, msk, 32);
      lrow[r] = lrow[r] * alpha[r] + v;
    }

    // ---- C-layout P -> A-layout via LDS ----
#pragma unroll
    for (int sub = 0; sub < 2; ++sub)
#pragma unroll
      for (int r = 0; r < 8; ++r) {
        int m = r + 8 * half;
        Plds[m * 32 + sub * 16 + n] = f2bf(sc[sub][r]);
      }
    __syncthreads();
    AFrag pa = load_a(Plds, 32, 0, 0, lane);

    // ---- O += P V  (V^T storage: contiguous keys per dim) ----
#pragma unroll
    for (int nb = 0; nb < 4; ++nb) {
      BFrag bv = load_bT(Vh, S, nb * 16, kt, lane);
      o[nb] = wmma_bf16(pa, bv, o[nb]);
    }
    __syncthreads();
  }

  // ---- normalize + store Ob[s][h*64 + d] ----
#pragma unroll
  for (int nb = 0; nb < 4; ++nb) {
    int d = nb * 16 + n;
#pragma unroll
    for (int r = 0; r < 8; ++r) {
      int s = q0 + r + 8 * half;
      Ob[(size_t)s * NQD + h * D + d] = f2bf(o[nb][r] / lrow[r]);
    }
  }
}

// ---------------- output projection: out = Ob @ Wo + bo (fp32 out) --------
__global__ __launch_bounds__(32)
void out_proj_kernel(const unsigned short* __restrict__ Ob,   // [S][NQD]
                     const unsigned short* __restrict__ Wot,  // [EMB][NQD]
                     const float* __restrict__ bo,
                     float* __restrict__ out) {               // [S][EMB]
  int lane = threadIdx.x;
  int s0 = blockIdx.x * 16;
  int c0 = blockIdx.y * 64;

  v8f acc[4] = {};
  for (int k0 = 0; k0 < NQD; k0 += 32) {
    AFrag a = load_a(Ob, NQD, s0, k0, lane);
#pragma unroll
    for (int nb = 0; nb < 4; ++nb) {
      BFrag b = load_bT(Wot, NQD, c0 + nb * 16, k0, lane);
      acc[nb] = wmma_bf16(a, b, acc[nb]);
    }
  }

  int n = lane & 15, half = lane >> 4;
#pragma unroll
  for (int nb = 0; nb < 4; ++nb) {
    int c = c0 + nb * 16 + n;
    float bb = bo[c];
#pragma unroll
    for (int r = 0; r < 8; ++r) {
      int s = s0 + r + 8 * half;
      out[(size_t)s * EMB + c] = acc[nb][r] + bb;
    }
  }
}

// ---------------- host launcher ----------------
extern "C" void kernel_launch(void* const* d_in, const int* in_sizes, int n_in,
                              void* d_out, int out_size, void* d_ws, size_t ws_size,
                              hipStream_t stream) {
  const float* x  = (const float*)d_in[0];
  const float* Wq = (const float*)d_in[1];
  const float* bq = (const float*)d_in[2];
  const float* Wk = (const float*)d_in[3];
  const float* bk = (const float*)d_in[4];
  const float* Wv = (const float*)d_in[5];
  const float* bv = (const float*)d_in[6];
  const float* Wo = (const float*)d_in[7];
  const float* bo = (const float*)d_in[8];
  float* out = (float*)d_out;

  char* ws = (char*)d_ws;
  size_t off = 0;
  auto carve = [&](size_t elems) -> unsigned short* {
    unsigned short* p = (unsigned short*)(ws + off);
    off += ((elems * sizeof(unsigned short)) + 255) & ~(size_t)255;
    return p;
  };
  unsigned short* Xbf  = carve((size_t)S * EMB);     // x in bf16
  unsigned short* Wqt  = carve((size_t)NQD * EMB);   // Wq^T
  unsigned short* Wkt  = carve((size_t)NKVD * EMB);  // Wk^T
  unsigned short* Wvt  = carve((size_t)NKVD * EMB);  // Wv^T
  unsigned short* Wot  = carve((size_t)EMB * NQD);   // Wo^T
  unsigned short* Qb   = carve((size_t)NQ * S * D);  // roped Q, head-major
  unsigned short* Kb   = carve((size_t)NKV * S * D); // roped K, head-major
  unsigned short* Vtb  = carve((size_t)NKV * D * S); // V transposed
  unsigned short* Obuf = carve((size_t)S * NQD);     // attention output
  (void)ws_size; (void)in_sizes; (void)n_in; (void)out_size;

  // 1. precision casts / transposes
  {
    int nx = S * EMB;
    cast_f32_bf16_kernel<<<(nx + 255) / 256, 256, 0, stream>>>(x, Xbf, nx);
    int nq = EMB * NQD;
    transpose_cast_kernel<<<(nq + 255) / 256, 256, 0, stream>>>(Wq, Wqt, EMB, NQD);
    int nk = EMB * NKVD;
    transpose_cast_kernel<<<(nk + 255) / 256, 256, 0, stream>>>(Wk, Wkt, EMB, NKVD);
    transpose_cast_kernel<<<(nk + 255) / 256, 256, 0, stream>>>(Wv, Wvt, EMB, NKVD);
    int no = NQD * EMB;
    transpose_cast_kernel<<<(no + 255) / 256, 256, 0, stream>>>(Wo, Wot, NQD, EMB);
  }

  // 2. fused QKV projection + bias + RoPE (WMMA)
  qkv_rope_kernel<<<dim3(S / 16, NQ + 2 * NKV), dim3(32), 0, stream>>>(
      Xbf, Wqt, bq, Wkt, bk, Wvt, bv, Qb, Kb, Vtb);

  // 3. causal flash attention (WMMA)
  attn_kernel<<<dim3(S / 16, NQ), dim3(32), 0, stream>>>(Qb, Kb, Vtb, Obuf);

  // 4. output projection (WMMA)
  out_proj_kernel<<<dim3(S / 16, EMB / 64), dim3(32), 0, stream>>>(Obuf, Wot, bo, out);
}
